// DecoderClassification_78408922956175
// MI455X (gfx1250) — matrix operand order repaired and checked by reading, not compiled, not run on hardware
//
#include <hip/hip_runtime.h>
#include <hip/hip_bf16.h>

// Problem constants (from the reference)
#define B_  32
#define T_  2048
#define H_  2048
#define S_  128
#define L_  8
#define M_  (B_ * S_)   // 4096 rows (spans)
#define K_  H_          // 2048 reduction dim
#define N_  H_          // 2048 hidden dim

typedef __attribute__((ext_vector_type(16))) __bf16         v16bf;
typedef __attribute__((ext_vector_type(8)))  float          v8f;
typedef __attribute__((ext_vector_type(4)))  float          v4f;
typedef __attribute__((ext_vector_type(8)))  unsigned short u16x8;
typedef __attribute__((ext_vector_type(16))) unsigned short u16x16;

union BfFrag { u16x16 u; v16bf v; };

// float -> bf16 (round-to-nearest-even), raw bits
static __device__ __forceinline__ unsigned short f2bf(float f) {
    unsigned int u = __float_as_uint(f);
    u += 0x7FFFu + ((u >> 16) & 1u);
    return (unsigned short)(u >> 16);
}
static __device__ __forceinline__ float bf2f(unsigned short b) {
    return __uint_as_float(((unsigned int)b) << 16);
}

// -------------------------------------------------------------------------
// Kernel 1: W1 [K,N] f32 row-major -> W1T [N,K] bf16 row-major (tiled transpose)
// -------------------------------------------------------------------------
__global__ void w1_transpose_kernel(const float* __restrict__ W1,
                                    unsigned short* __restrict__ W1T) {
    __shared__ float tile[32][33];
    const int n0 = blockIdx.x * 32;   // N block
    const int k0 = blockIdx.y * 32;   // K block
    const int tx = threadIdx.x;       // 0..31
    const int ty = threadIdx.y;       // 0..7
    #pragma unroll
    for (int r = 0; r < 32; r += 8)
        tile[ty + r][tx] = W1[(size_t)(k0 + ty + r) * N_ + (n0 + tx)];
    __syncthreads();
    #pragma unroll
    for (int r = 0; r < 32; r += 8)
        W1T[(size_t)(n0 + ty + r) * K_ + (k0 + tx)] = f2bf(tile[tx][ty + r]);
}

// -------------------------------------------------------------------------
// Kernel 2: ragged span mean pooling, h f32 [B,T,H] -> pooled bf16 [M,H]
// One block per span; 256 threads; each thread owns two float4 slots per row
// (2 x global_load_b128, fully coalesced). This kernel carries the dominant
// HBM traffic (~1.1 GB), so it is vectorized aggressively.
// Also emits the labels output (second element of the reference tuple).
// -------------------------------------------------------------------------
__global__ void span_pool_kernel(const float* __restrict__ h,
                                 const int* __restrict__ y,
                                 unsigned short* __restrict__ pooled,
                                 float* __restrict__ out_labels) {
    const int span  = blockIdx.x;            // 0..M-1
    const int b     = span / S_;
    const int start = y[span * 3 + 0];
    const int end   = y[span * 3 + 1];
    const int label = y[span * 3 + 2];
    const int tid   = threadIdx.x;           // 0..255

    v4f acc0 = (v4f)0.0f, acc1 = (v4f)0.0f;

    const float* hb = h + ((size_t)b * T_ + start) * H_;
    const int len = end - start;
    #pragma unroll 2
    for (int t = 0; t < len; ++t) {
        const v4f* row4 = (const v4f*)(hb + (size_t)t * H_);
        acc0 += row4[tid];            // channels 4*tid .. 4*tid+3
        acc1 += row4[tid + 256];      // channels 1024 + 4*tid ..
    }
    const float inv = 1.0f / (float)len;
    unsigned short* dst = pooled + (size_t)span * H_;
    ushort4 o0, o1;
    o0.x = f2bf(acc0[0] * inv); o0.y = f2bf(acc0[1] * inv);
    o0.z = f2bf(acc0[2] * inv); o0.w = f2bf(acc0[3] * inv);
    o1.x = f2bf(acc1[0] * inv); o1.y = f2bf(acc1[1] * inv);
    o1.z = f2bf(acc1[2] * inv); o1.w = f2bf(acc1[3] * inv);
    *(ushort4*)(dst + 4 * tid)        = o0;
    *(ushort4*)(dst + 1024 + 4 * tid) = o1;

    if (tid == 0) out_labels[span] = (float)label;
}

// -------------------------------------------------------------------------
// Kernel 3: hid = relu(pooled @ W1 + b1), bf16 WMMA with f32 accumulation.
//   A = pooled   bf16 [M,K] row-major
//   B = W1T      bf16 [N,K] row-major  (K contiguous -> 32B fragment loads)
// Block: 256 threads = 8 waves (wave32). Block tile 128(M) x 128(N).
// Waves arranged 2(M) x 4(N); wave tile 64 x 32 = 4x2 fragments of 16x16.
// Software-pipelined: ping-pong fragment sets so the loads for K-chunk k+32
// are in flight while the WMMAs for chunk k execute (hides L2 latency).
// -------------------------------------------------------------------------
struct Frags {
    v16bf a[4];
    v16bf b[2];
};

static __device__ __forceinline__ void load_frags(
        const unsigned short* const (&aRow)[4],
        const unsigned short* const (&bRow)[2],
        int k, int half, Frags& f) {
    // A fragment (16x32 bf16): per-lane elems 0..7 = K[8*half .. +7],
    // elems 8..15 = K[16 + 8*half .. +7]   (ISA 7.12.2 layout)
    #pragma unroll
    for (int mf = 0; mf < 4; ++mf) {
        const u16x8 lo = *(const u16x8*)(aRow[mf] + k + half * 8);
        const u16x8 hi = *(const u16x8*)(aRow[mf] + k + half * 8 + 16);
        BfFrag t;
        #pragma unroll
        for (int i = 0; i < 8; ++i) { t.u[i] = lo[i]; t.u[i + 8] = hi[i]; }
        f.a[mf] = t.v;
    }
    // B fragment (32x16 bf16): per-lane elems 0..7 = K[8*half .. +7],
    // elems 8..15 = K[16 + 8*half .. +7]
    #pragma unroll
    for (int nf = 0; nf < 2; ++nf) {
        const u16x8 lo = *(const u16x8*)(bRow[nf] + k + half * 8);
        const u16x8 hi = *(const u16x8*)(bRow[nf] + k + half * 8 + 16);
        BfFrag t;
        #pragma unroll
        for (int i = 0; i < 8; ++i) { t.u[i] = lo[i]; t.u[i + 8] = hi[i]; }
        f.b[nf] = t.v;
    }
}

static __device__ __forceinline__ void mma_step(const Frags& f, v8f (&acc)[4][2]) {
    #pragma unroll
    for (int mf = 0; mf < 4; ++mf)
        #pragma unroll
        for (int nf = 0; nf < 2; ++nf)
            acc[mf][nf] = __builtin_amdgcn_wmma_f32_16x16x32_bf16(
                false, f.a[mf], false, f.b[nf],
                (short)0, acc[mf][nf], false, false);
}

__global__ void __launch_bounds__(256)
gemm1_wmma_kernel(const unsigned short* __restrict__ A,
                  const unsigned short* __restrict__ Bt,
                  const float* __restrict__ b1,
                  unsigned short* __restrict__ hid) {
    const int lane = threadIdx.x & 31;
    const int wave = threadIdx.x >> 5;       // 0..7
    const int wm   = wave & 1;               // wave row (M)
    const int wn   = wave >> 1;              // wave col (N)
    const int rowBase = blockIdx.y * 128 + wm * 64;
    const int colBase = blockIdx.x * 128 + wn * 32;
    const int half = lane >> 4;              // 0 or 1
    const int l16  = lane & 15;

    v8f acc[4][2];
    #pragma unroll
    for (int mf = 0; mf < 4; ++mf)
        #pragma unroll
        for (int nf = 0; nf < 2; ++nf)
            acc[mf][nf] = (v8f)0.0f;

    const unsigned short* aRow[4];
    #pragma unroll
    for (int mf = 0; mf < 4; ++mf)
        aRow[mf] = A + (size_t)(rowBase + mf * 16 + l16) * K_;
    const unsigned short* bRow[2];
    #pragma unroll
    for (int nf = 0; nf < 2; ++nf)
        bRow[nf] = Bt + (size_t)(colBase + nf * 16 + l16) * K_;

    Frags f0, f1;
    load_frags(aRow, bRow, 0, half, f0);

    // steady state: two 32-K chunks per loop body, ping-pong buffers
    #pragma unroll 1
    for (int kk = 0; kk < K_ - 64; kk += 64) {
        load_frags(aRow, bRow, kk + 32, half, f1);
        mma_step(f0, acc);
        load_frags(aRow, bRow, kk + 64, half, f0);
        mma_step(f1, acc);
    }
    // tail: chunks K_-64 (already in f0) and K_-32
    load_frags(aRow, bRow, K_ - 32, half, f1);
    mma_step(f0, acc);
    mma_step(f1, acc);

    // Epilogue: D layout (16x16 f32): elem r -> M = r + 8*half, N = l16.
    #pragma unroll
    for (int nf = 0; nf < 2; ++nf) {
        const int col  = colBase + nf * 16 + l16;
        const float bias = b1[col];
        #pragma unroll
        for (int mf = 0; mf < 4; ++mf) {
            #pragma unroll
            for (int r = 0; r < 8; ++r) {
                const int row = rowBase + mf * 16 + half * 8 + r;
                float v = acc[mf][nf][r] + bias;
                v = v > 0.0f ? v : 0.0f;
                hid[(size_t)row * N_ + col] = f2bf(v);
            }
        }
    }
}

// -------------------------------------------------------------------------
// Kernel 4: logits = hid @ W2 + b2.  N=8 is too narrow for WMMA; this is a
// bandwidth-bound reduction. One block (256 threads) per output row; each
// thread covers exactly 8 contiguous K values (one u16x8 load of hid), then
// an LDS tree reduction.
// -------------------------------------------------------------------------
__global__ void gemm2_kernel(const unsigned short* __restrict__ hid,
                             const float* __restrict__ W2,
                             const float* __restrict__ b2,
                             float* __restrict__ logits) {
    __shared__ float red[256 * L_];
    const int row = blockIdx.x;
    const int tid = threadIdx.x;
    const int k0  = tid * 8;                 // 256*8 = 2048 = H_

    const u16x8 hv8 = *(const u16x8*)(hid + (size_t)row * H_ + k0);

    float p[L_];
    #pragma unroll
    for (int l = 0; l < L_; ++l) p[l] = 0.0f;

    #pragma unroll
    for (int j = 0; j < 8; ++j) {
        const float hv = bf2f(hv8[j]);
        const v4f w0 = *(const v4f*)(W2 + (size_t)(k0 + j) * L_);
        const v4f w1 = *(const v4f*)(W2 + (size_t)(k0 + j) * L_ + 4);
        #pragma unroll
        for (int l = 0; l < 4; ++l) p[l]     += hv * w0[l];
        #pragma unroll
        for (int l = 0; l < 4; ++l) p[l + 4] += hv * w1[l];
    }
    #pragma unroll
    for (int l = 0; l < L_; ++l) red[tid * L_ + l] = p[l];
    __syncthreads();

    for (int off = 128; off > 0; off >>= 1) {
        if (tid < off) {
            #pragma unroll
            for (int l = 0; l < L_; ++l)
                red[tid * L_ + l] += red[(tid + off) * L_ + l];
        }
        __syncthreads();
    }
    if (tid < L_)
        logits[(size_t)row * L_ + tid] = red[tid] + b2[tid];
}

// -------------------------------------------------------------------------
// Launch
// -------------------------------------------------------------------------
extern "C" void kernel_launch(void* const* d_in, const int* in_sizes, int n_in,
                              void* d_out, int out_size, void* d_ws, size_t ws_size,
                              hipStream_t stream) {
    const float* h  = (const float*)d_in[0];   // [B,T,H] f32
    const int*   y  = (const int*)d_in[1];     // [B,S,3] i32
    // d_in[2] = x, unused by the reference computation
    const float* W1 = (const float*)d_in[3];   // [H,H] f32
    const float* b1 = (const float*)d_in[4];   // [H]   f32
    const float* W2 = (const float*)d_in[5];   // [H,L] f32
    const float* b2 = (const float*)d_in[6];   // [L]   f32

    float* out = (float*)d_out;                // logits [M,L] then labels [M]

    // Workspace layout (bf16 as raw u16):
    //   pooled [M,K]  : 16 MB
    //   W1T    [N,K]  :  8 MB
    //   hid    [M,N]  : 16 MB   (all L2-resident: 40 MB << 192 MB)
    char* ws = (char*)d_ws;
    unsigned short* pooled = (unsigned short*)ws;
    unsigned short* W1T    = (unsigned short*)(ws + (size_t)M_ * K_ * 2);
    unsigned short* hid    = (unsigned short*)(ws + (size_t)M_ * K_ * 2
                                                  + (size_t)N_ * K_ * 2);

    w1_transpose_kernel<<<dim3(N_ / 32, K_ / 32), dim3(32, 8), 0, stream>>>(W1, W1T);
    span_pool_kernel<<<M_, 256, 0, stream>>>(h, y, pooled, out + (size_t)M_ * L_);
    gemm1_wmma_kernel<<<dim3(N_ / 128, M_ / 128), 256, 0, stream>>>(pooled, W1T, b1, hid);
    gemm2_kernel<<<M_, 256, 0, stream>>>(hid, W2, b2, out);
}
